// MultiHeadAttention_43078521979238
// MI455X (gfx1250) — matrix operand order repaired and checked by reading, not compiled, not run on hardware
//
#include <hip/hip_runtime.h>
#include <hip/hip_bf16.h>

// ---------------- problem constants ----------------
constexpr int Bn = 2;          // batch
constexpr int T  = 2048;       // sequence
constexpr int Cc = 1024;       // channels
constexpr int H  = 16;         // heads
constexpr int Dh = 64;         // head dim
constexpr int HD = H * Dh;     // 1024

typedef __bf16 bf16;
typedef __attribute__((ext_vector_type(8)))  __bf16 bf16x8;
typedef __attribute__((ext_vector_type(16))) __bf16 bf16x16;
typedef __attribute__((ext_vector_type(8)))  float  f32x8;

// ---------------- WMMA helpers ----------------
__device__ __forceinline__ f32x8 wmma_bf16(bf16x16 a, bf16x16 b, f32x8 c) {
  return __builtin_amdgcn_wmma_f32_16x16x32_bf16(
      false, a, false, b, (short)0, c, false, false);
}

// A-fragment (16x32): lane row M=lane%16, K = 16*(lane/16)+i.
__device__ __forceinline__ bf16x16 frag_a(const bf16* p) {
  bf16x8 lo = *reinterpret_cast<const bf16x8*>(p);
  bf16x8 hi = *reinterpret_cast<const bf16x8*>(p + 8);
  bf16x16 f;
#pragma unroll
  for (int i = 0; i < 8; ++i) { f[i] = lo[i]; f[i + 8] = hi[i]; }
  return f;
}

// B-fragment (32x16), source row-major over K: 32 contiguous bytes.
__device__ __forceinline__ bf16x16 frag_b(const bf16* p) {
  return *reinterpret_cast<const bf16x16*>(p);
}

// LDS byte offset of a __shared__ object
__device__ __forceinline__ unsigned lds_off(const void* p) {
  return (unsigned)(unsigned long long)
      (__attribute__((address_space(3))) const void*)p;
}

// Execution-only workgroup barrier (no counter drain; also a compiler
// scheduling fence so prefetched loads cannot sink across it).
__device__ __forceinline__ void wg_barrier() {
  asm volatile("s_barrier_signal -1\n\ts_barrier_wait -1" ::: "memory");
}

// ---------------- conversion / transpose kernels ----------------
__global__ void cvt_f32_bf16(const float* __restrict__ in, bf16* __restrict__ out, int n) {
  int i = blockIdx.x * blockDim.x + threadIdx.x;
  if (i < n) out[i] = (bf16)in[i];
}

// W[h][c][d] -> Wt[h][d][c]
__global__ void transpose_w(const float* __restrict__ W, bf16* __restrict__ Wt) {
  int i = blockIdx.x * blockDim.x + threadIdx.x;
  if (i < H * Cc * Dh) {
    int d = i % Dh;
    int c = (i / Dh) % Cc;
    int h = i / (Cc * Dh);
    Wt[((size_t)h * Dh + d) * Cc + c] = (bf16)W[i];
  }
}

// Wp[i][c] -> Wpt[c][i]
__global__ void transpose_wp(const float* __restrict__ Wp, bf16* __restrict__ Wpt) {
  int j = blockIdx.x * blockDim.x + threadIdx.x;
  if (j < HD * Cc) {
    int c = j % Cc;
    int i = j / Cc;
    Wpt[(size_t)c * HD + i] = (bf16)Wp[j];
  }
}

// ---------------- kernel 1: QKV projection (one matrix per grid.y) ----------------
// 8 waves x 32 rows = 256 rows/block. Weight tile (64x32 bf16 = 4KB) staged in
// LDS via async DMA, 3-deep. Fragments for chunk i+1 are prefetched into
// registers during chunk i's WMMA burst. Steady-state loop is 30 iterations,
// unrolled 3x so all circular-buffer indices are compile-time constants and
// the hot loop is branch-free; the last two chunks are peeled.
__global__ void __launch_bounds__(256)
qkv_kernel(const bf16* __restrict__ xb,
           const bf16* __restrict__ wq_t,
           const bf16* __restrict__ wk_t,
           const bf16* __restrict__ wv_t,
           bf16* __restrict__ qb,   // [B,H,T,Dh]
           bf16* __restrict__ kb,   // [B,H,T,Dh]
           bf16* __restrict__ vt) { // [B,H,Dh,T]
  __shared__ alignas(32) bf16 wbuf[3][64 * 32];   // 12 KB

  const int tid  = threadIdx.x;
  const int lane = tid & 31;
  const int wave = tid >> 5;
  const int lm   = lane & 15;
  const int hf   = lane >> 4;
  const int mat  = blockIdx.y;                 // 0=Q 1=K 2=V

  const int rowBlocks = T / 256;               // 8
  const int bh     = blockIdx.x / rowBlocks;
  const int rowBlk = blockIdx.x % rowBlocks;
  const int b = bh / H, h = bh % H;
  const int mbase = rowBlk * 256 + wave * 32;  // 2 row tiles per wave

  const bf16* A = xb + (size_t)b * T * Cc;
  const bf16* wsel = (mat == 0) ? wq_t : (mat == 1) ? wk_t : wv_t;
  const unsigned long long wsrc =
      (unsigned long long)(wsel + (size_t)h * Dh * Cc);

  // async tile fill: 256 threads x 16B = 64x32 bf16 tile (1 instr/tile)
  const int frow = tid >> 2;            // 0..63 (d)
  const int fcol = (tid & 3) * 8;
  auto issue = [&](int buf, int kc) {
    unsigned ldsa = lds_off(&wbuf[buf][frow * 32 + fcol]);
    unsigned goff = (unsigned)(((size_t)frow * Cc + kc + fcol) * sizeof(bf16));
    asm volatile("global_load_async_to_lds_b128 %0, %1, %2"
                 :: "v"(ldsa), "v"(goff), "s"(wsrc) : "memory");
  };

  f32x8 acc[2][4];
  const f32x8 zero = {0.f,0.f,0.f,0.f,0.f,0.f,0.f,0.f};
#pragma unroll
  for (int rt = 0; rt < 2; ++rt)
#pragma unroll
    for (int nt = 0; nt < 4; ++nt) acc[rt][nt] = zero;

  const bf16* arow[2];
#pragma unroll
  for (int rt = 0; rt < 2; ++rt)
    arow[rt] = A + (size_t)(mbase + rt * 16 + lm) * Cc + 16 * hf;

  issue(0, 0);
  issue(1, 32);
  asm volatile("s_wait_asynccnt 0x1" ::: "memory");   // tile 0 complete
  wg_barrier();

  bf16x16 acur[2], bcur[4];
#pragma unroll
  for (int rt = 0; rt < 2; ++rt) acur[rt] = frag_a(arow[rt]);
#pragma unroll
  for (int nt = 0; nt < 4; ++nt)
    bcur[nt] = frag_b(&wbuf[0][(nt * 16 + lm) * 32 + 16 * hf]);

  constexpr int NCH = Cc / 32;       // 32 chunks; steady state = 30 (div by 3)
#pragma unroll 3
  for (int i = 0; i < NCH - 2; ++i) {
    const int kc = i * 32;
    issue((i + 2) % 3, kc + 64);                       // const under unroll 3
    asm volatile("s_wait_asynccnt 0x1" ::: "memory");  // tiles <= i+1 done
    wg_barrier();

    bf16x16 anext[2], bnext[4];
#pragma unroll
    for (int rt = 0; rt < 2; ++rt) anext[rt] = frag_a(arow[rt] + kc + 32);
    const int nb = (i + 1) % 3;                        // const under unroll 3
#pragma unroll
    for (int nt = 0; nt < 4; ++nt)
      bnext[nt] = frag_b(&wbuf[nb][(nt * 16 + lm) * 32 + 16 * hf]);

#pragma unroll
    for (int rt = 0; rt < 2; ++rt)
#pragma unroll
      for (int nt = 0; nt < 4; ++nt)
        acc[rt][nt] = wmma_bf16(acur[rt], bcur[nt], acc[rt][nt]);

    wg_barrier();
#pragma unroll
    for (int rt = 0; rt < 2; ++rt) acur[rt] = anext[rt];
#pragma unroll
    for (int nt = 0; nt < 4; ++nt) bcur[nt] = bnext[nt];
  }

  {                                   // chunk NCH-2: last tile already issued
    asm volatile("s_wait_asynccnt 0x0" ::: "memory");  // tile NCH-1 complete
    wg_barrier();
    bf16x16 anext[2], bnext[4];
#pragma unroll
    for (int rt = 0; rt < 2; ++rt)
      anext[rt] = frag_a(arow[rt] + (NCH - 1) * 32);
#pragma unroll
    for (int nt = 0; nt < 4; ++nt)
      bnext[nt] = frag_b(&wbuf[(NCH - 1) % 3][(nt * 16 + lm) * 32 + 16 * hf]);
#pragma unroll
    for (int rt = 0; rt < 2; ++rt)
#pragma unroll
      for (int nt = 0; nt < 4; ++nt)
        acc[rt][nt] = wmma_bf16(acur[rt], bcur[nt], acc[rt][nt]);
#pragma unroll
    for (int rt = 0; rt < 2; ++rt) acur[rt] = anext[rt];
#pragma unroll
    for (int nt = 0; nt < 4; ++nt) bcur[nt] = bnext[nt];
  }
#pragma unroll
  for (int rt = 0; rt < 2; ++rt)      // chunk NCH-1
#pragma unroll
    for (int nt = 0; nt < 4; ++nt)
      acc[rt][nt] = wmma_bf16(acur[rt], bcur[nt], acc[rt][nt]);

  // epilogue stores (uniform branch hoisted)
  if (mat == 2) {
#pragma unroll
    for (int rt = 0; rt < 2; ++rt)
#pragma unroll
      for (int nt = 0; nt < 4; ++nt)
#pragma unroll
        for (int r = 0; r < 8; ++r) {
          const int row = mbase + rt * 16 + r + 8 * hf;
          const int col = nt * 16 + lm;
          vt[((size_t)bh * Dh + col) * T + row] = (bf16)acc[rt][nt][r];
        }
  } else {
    bf16* dst = (mat == 0) ? qb : kb;
#pragma unroll
    for (int rt = 0; rt < 2; ++rt)
#pragma unroll
      for (int nt = 0; nt < 4; ++nt)
#pragma unroll
        for (int r = 0; r < 8; ++r) {
          const int row = mbase + rt * 16 + r + 8 * hf;
          const int col = nt * 16 + lm;
          dst[((size_t)bh * T + row) * Dh + col] = (bf16)acc[rt][nt][r];
        }
  }
}

// ---------------- kernel 2: causal flash attention ----------------
__device__ __forceinline__ void load_kfrags(const bf16* Kp, int kbase, int lm, int hf,
                                            bf16x16 out[2][2]) {
#pragma unroll
  for (int sub = 0; sub < 2; ++sub) {
    const bf16* krow = Kp + (size_t)(kbase + sub * 16 + lm) * Dh + 16 * hf;
    out[sub][0] = frag_b(krow);
    out[sub][1] = frag_b(krow + 32);
  }
}
__device__ __forceinline__ void load_vfrags(const bf16* Vt, int kbase, int lm, int hf,
                                            bf16x16 out[4]) {
#pragma unroll
  for (int nt = 0; nt < 4; ++nt)
    out[nt] = frag_b(Vt + (size_t)(nt * 16 + lm) * T + kbase + 16 * hf);
}

__global__ void __launch_bounds__(128)
attn_kernel(const bf16* __restrict__ qb,   // [B,H,T,Dh]
            const bf16* __restrict__ kb,   // [B,H,T,Dh]
            const bf16* __restrict__ vt,   // [B,H,Dh,T]
            bf16* __restrict__ ob) {       // [B,T,H*Dh]
  __shared__ alignas(32) bf16 pbuf[4][16][40];

  const int lane = threadIdx.x & 31;
  const int wave = threadIdx.x >> 5;
  const int lm   = lane & 15;
  const int hf   = lane >> 4;

  const int g     = blockIdx.x * 4 + wave;
  const int bh    = g >> 7;
  const int qt    = g & 127;
  const int qbase = qt * 16;
  const int b = bh / H, h = bh % H;

  const bf16* Qp = qb + ((size_t)bh * T + qbase) * Dh;
  const bf16* Kp = kb + (size_t)bh * T * Dh;
  const bf16* Vt = vt + (size_t)bh * Dh * T;

  const bf16x16 aq0 = frag_a(Qp + (size_t)lm * Dh + 0  + 16 * hf);
  const bf16x16 aq1 = frag_a(Qp + (size_t)lm * Dh + 32 + 16 * hf);

  const f32x8 zero = {0.f,0.f,0.f,0.f,0.f,0.f,0.f,0.f};
  f32x8 o[4];
#pragma unroll
  for (int nt = 0; nt < 4; ++nt) o[nt] = zero;

  float m[8], l[8];
#pragma unroll
  for (int r = 0; r < 8; ++r) { m[r] = -1e30f; l[r] = 0.f; }

  const int kend = qbase + 15;

  bf16x16 kf[2][2], vf[4];
  load_kfrags(Kp, 0, lm, hf, kf);
  load_vfrags(Vt, 0, lm, hf, vf);

  for (int kbase = 0; kbase <= kend; kbase += 32) {
    const bool hasn = (kbase + 32) <= kend;
    bf16x16 kn[2][2], vn[4];
    if (hasn) {                       // prefetch next key block
      load_kfrags(Kp, kbase + 32, lm, hf, kn);
      load_vfrags(Vt, kbase + 32, lm, hf, vn);
    }

    f32x8 s[2];
    float mloc[8];
#pragma unroll
    for (int r = 0; r < 8; ++r) mloc[r] = -1e30f;

#pragma unroll
    for (int sub = 0; sub < 2; ++sub) {
      const int key = kbase + sub * 16 + lm;
      f32x8 sv = zero;
      sv = wmma_bf16(aq0, kf[sub][0], sv);
      sv = wmma_bf16(aq1, kf[sub][1], sv);
#pragma unroll
      for (int r = 0; r < 8; ++r) {
        float x = sv[r] * 0.125f;
        const int qrow = qbase + r + 8 * hf;
        if (key > qrow) x = -1e30f;
        sv[r] = x;
        mloc[r] = fmaxf(mloc[r], x);
      }
      s[sub] = sv;
    }

#pragma unroll
    for (int r = 0; r < 8; ++r) {
      mloc[r] = fmaxf(mloc[r], __shfl_xor(mloc[r], 1, 32));
      mloc[r] = fmaxf(mloc[r], __shfl_xor(mloc[r], 2, 32));
      mloc[r] = fmaxf(mloc[r], __shfl_xor(mloc[r], 4, 32));
      mloc[r] = fmaxf(mloc[r], __shfl_xor(mloc[r], 8, 32));
    }

    float alpha[8];
#pragma unroll
    for (int r = 0; r < 8; ++r) {
      const float mn = fmaxf(m[r], mloc[r]);
      alpha[r] = __expf(m[r] - mn);
      m[r] = mn;
    }
#pragma unroll
    for (int nt = 0; nt < 4; ++nt)
#pragma unroll
      for (int r = 0; r < 8; ++r) o[nt][r] *= alpha[r];

    float rs[8];
#pragma unroll
    for (int r = 0; r < 8; ++r) rs[r] = 0.f;
#pragma unroll
    for (int sub = 0; sub < 2; ++sub)
#pragma unroll
      for (int r = 0; r < 8; ++r) {
        const float p = __expf(s[sub][r] - m[r]);
        rs[r] += p;
        pbuf[wave][r + 8 * hf][sub * 16 + lm] = (bf16)p;
      }
#pragma unroll
    for (int r = 0; r < 8; ++r) {
      rs[r] += __shfl_xor(rs[r], 1, 32);
      rs[r] += __shfl_xor(rs[r], 2, 32);
      rs[r] += __shfl_xor(rs[r], 4, 32);
      rs[r] += __shfl_xor(rs[r], 8, 32);
      l[r] = l[r] * alpha[r] + rs[r];
    }

    const bf16x16 pf = frag_a(&pbuf[wave][lm][16 * hf]);  // LDS transpose read
#pragma unroll
    for (int nt = 0; nt < 4; ++nt) o[nt] = wmma_bf16(pf, vf[nt], o[nt]);

    if (hasn) {
#pragma unroll
      for (int sub = 0; sub < 2; ++sub) { kf[sub][0] = kn[sub][0]; kf[sub][1] = kn[sub][1]; }
#pragma unroll
      for (int nt = 0; nt < 4; ++nt) vf[nt] = vn[nt];
    }
  }

#pragma unroll
  for (int r = 0; r < 8; ++r) {
    const float inv = 1.0f / l[r];
    const int row = qbase + r + 8 * hf;
#pragma unroll
    for (int nt = 0; nt < 4; ++nt) {
      const int col = h * Dh + nt * 16 + lm;
      ob[((size_t)b * T + row) * HD + col] = (bf16)(o[nt][r] * inv);
    }
  }
}

// ---------------- kernel 3: output projection ----------------
// Same peeled, branch-free pipeline as qkv_kernel.
__global__ void __launch_bounds__(256)
proj_kernel(const bf16* __restrict__ ob,
            const bf16* __restrict__ wpt,   // [C][HD]
            float* __restrict__ out) {
  __shared__ alignas(32) bf16 wbuf[3][64 * 32];   // 12 KB

  const int tid  = threadIdx.x;
  const int lane = tid & 31;
  const int wave = tid >> 5;
  const int lm   = lane & 15;
  const int hf   = lane >> 4;

  const int nTiles  = Cc / 64;                   // 16
  const int colTile = blockIdx.x % nTiles;
  const int rowBlk  = blockIdx.x / nTiles;
  const int mbase = rowBlk * 256 + wave * 32;
  const int nbase = colTile * 64;

  const unsigned long long wsrc =
      (unsigned long long)(wpt + (size_t)nbase * HD);

  const int frow = tid >> 2;
  const int fcol = (tid & 3) * 8;
  auto issue = [&](int buf, int kc) {
    unsigned ldsa = lds_off(&wbuf[buf][frow * 32 + fcol]);
    unsigned goff = (unsigned)(((size_t)frow * HD + kc + fcol) * sizeof(bf16));
    asm volatile("global_load_async_to_lds_b128 %0, %1, %2"
                 :: "v"(ldsa), "v"(goff), "s"(wsrc) : "memory");
  };

  f32x8 acc[2][4];
  const f32x8 zero = {0.f,0.f,0.f,0.f,0.f,0.f,0.f,0.f};
#pragma unroll
  for (int rt = 0; rt < 2; ++rt)
#pragma unroll
    for (int nt = 0; nt < 4; ++nt) acc[rt][nt] = zero;

  const bf16* arow[2];
#pragma unroll
  for (int rt = 0; rt < 2; ++rt)
    arow[rt] = ob + (size_t)(mbase + rt * 16 + lm) * HD + 16 * hf;

  issue(0, 0);
  issue(1, 32);
  asm volatile("s_wait_asynccnt 0x1" ::: "memory");
  wg_barrier();

  bf16x16 acur[2], bcur[4];
#pragma unroll
  for (int rt = 0; rt < 2; ++rt) acur[rt] = frag_a(arow[rt]);
#pragma unroll
  for (int nt = 0; nt < 4; ++nt)
    bcur[nt] = frag_b(&wbuf[0][(nt * 16 + lm) * 32 + 16 * hf]);

  constexpr int NCH = HD / 32;       // 32
#pragma unroll 3
  for (int i = 0; i < NCH - 2; ++i) {
    const int kc = i * 32;
    issue((i + 2) % 3, kc + 64);
    asm volatile("s_wait_asynccnt 0x1" ::: "memory");
    wg_barrier();

    bf16x16 anext[2], bnext[4];
#pragma unroll
    for (int rt = 0; rt < 2; ++rt) anext[rt] = frag_a(arow[rt] + kc + 32);
    const int nb = (i + 1) % 3;
#pragma unroll
    for (int nt = 0; nt < 4; ++nt)
      bnext[nt] = frag_b(&wbuf[nb][(nt * 16 + lm) * 32 + 16 * hf]);

#pragma unroll
    for (int rt = 0; rt < 2; ++rt)
#pragma unroll
      for (int nt = 0; nt < 4; ++nt)
        acc[rt][nt] = wmma_bf16(acur[rt], bcur[nt], acc[rt][nt]);

    wg_barrier();
#pragma unroll
    for (int rt = 0; rt < 2; ++rt) acur[rt] = anext[rt];
#pragma unroll
    for (int nt = 0; nt < 4; ++nt) bcur[nt] = bnext[nt];
  }

  {                                   // chunk NCH-2
    asm volatile("s_wait_asynccnt 0x0" ::: "memory");
    wg_barrier();
    bf16x16 anext[2], bnext[4];
#pragma unroll
    for (int rt = 0; rt < 2; ++rt)
      anext[rt] = frag_a(arow[rt] + (NCH - 1) * 32);
#pragma unroll
    for (int nt = 0; nt < 4; ++nt)
      bnext[nt] = frag_b(&wbuf[(NCH - 1) % 3][(nt * 16 + lm) * 32 + 16 * hf]);
#pragma unroll
    for (int rt = 0; rt < 2; ++rt)
#pragma unroll
      for (int nt = 0; nt < 4; ++nt)
        acc[rt][nt] = wmma_bf16(acur[rt], bcur[nt], acc[rt][nt]);
#pragma unroll
    for (int rt = 0; rt < 2; ++rt) acur[rt] = anext[rt];
#pragma unroll
    for (int nt = 0; nt < 4; ++nt) bcur[nt] = bnext[nt];
  }
#pragma unroll
  for (int rt = 0; rt < 2; ++rt)      // chunk NCH-1
#pragma unroll
    for (int nt = 0; nt < 4; ++nt)
      acc[rt][nt] = wmma_bf16(acur[rt], bcur[nt], acc[rt][nt]);

#pragma unroll
  for (int rt = 0; rt < 2; ++rt)
#pragma unroll
    for (int nt = 0; nt < 4; ++nt)
#pragma unroll
      for (int r = 0; r < 8; ++r) {
        const int row = mbase + rt * 16 + r + 8 * hf;
        const int col = nbase + nt * 16 + lm;
        out[(size_t)row * Cc + col] = acc[rt][nt][r];
      }
}

// ---------------- launcher ----------------
extern "C" void kernel_launch(void* const* d_in, const int* in_sizes, int n_in,
                              void* d_out, int out_size, void* d_ws, size_t ws_size,
                              hipStream_t stream) {
  (void)in_sizes; (void)n_in; (void)out_size; (void)ws_size;

  const float* x  = (const float*)d_in[0];
  const float* Wq = (const float*)d_in[1];
  const float* Wk = (const float*)d_in[2];
  const float* Wv = (const float*)d_in[3];
  const float* Wp = (const float*)d_in[4];
  float* out = (float*)d_out;

  constexpr size_t XB_N  = (size_t)Bn * T * Cc;
  constexpr size_t W_N   = (size_t)H * Cc * Dh;
  constexpr size_t WP_N  = (size_t)HD * Cc;
  constexpr size_t QKV_N = (size_t)Bn * H * T * Dh;
  bf16* ws   = (bf16*)d_ws;
  bf16* xb   = ws;
  bf16* wq_t = xb + XB_N;
  bf16* wk_t = wq_t + W_N;
  bf16* wv_t = wk_t + W_N;
  bf16* wp_t = wv_t + W_N;
  bf16* qb   = wp_t + WP_N;
  bf16* kb   = qb + QKV_N;
  bf16* vt   = kb + QKV_N;
  bf16* ob   = vt + QKV_N;        // ~48 MB total

  cvt_f32_bf16<<<(XB_N + 255) / 256, 256, 0, stream>>>(x, xb, (int)XB_N);
  transpose_w <<<(W_N + 255) / 256, 256, 0, stream>>>(Wq, wq_t);
  transpose_w <<<(W_N + 255) / 256, 256, 0, stream>>>(Wk, wk_t);
  transpose_w <<<(W_N + 255) / 256, 256, 0, stream>>>(Wv, wv_t);
  transpose_wp<<<(WP_N + 255) / 256, 256, 0, stream>>>(Wp, wp_t);

  dim3 qkvGrid(Bn * H * (T / 256), 3);
  qkv_kernel<<<qkvGrid, 256, 0, stream>>>(xb, wq_t, wk_t, wv_t, qb, kb, vt);
  attn_kernel<<<Bn * H * (T / 16) / 4, 128, 0, stream>>>(qb, kb, vt, ob);
  proj_kernel<<<(Bn * T / 256) * (Cc / 64), 256, 0, stream>>>(ob, wp_t, out);
}
